// RNN_Dale_Sum_83476984365508
// MI455X (gfx1250) — matrix operand order repaired and checked by reading, not compiled, not run on hardware
//
#include <hip/hip_runtime.h>
#include <math.h>

#define NU 2048   // n_units
#define NI 16     // n_inputs
#define NB 64     // batch
#define NT 512    // timesteps
#define NBLK 64   // persistent blocks (one per 2 N-tiles)

typedef __attribute__((ext_vector_type(16))) __bf16 v16bf;
typedef __attribute__((ext_vector_type(8)))  float  v8f;
typedef __attribute__((ext_vector_type(4)))  int    v4i;   // plain 16B vector (not HIP class)

// explicit global-address-space pointers: force global_* (not flat_*) encodings
typedef __attribute__((address_space(1))) const v16bf* gv16c;
typedef __attribute__((address_space(1))) const float* gfc;
typedef __attribute__((address_space(1))) float*       gf;
typedef __attribute__((address_space(1))) __bf16*      gbf;
typedef __attribute__((address_space(1))) const v4i*   gv4c;

// ---------------------------------------------------------------------------
// Prep 1: Wsum = a*|Wd|*sdale[col] + (1-a)*Wn, cast to bf16, stored pre-
// swizzled into WMMA B-fragment order:
//   flat index = ((nt*64 + kc)*32 + lane)*16 + e,  lane = (k32>>4)*16 + (n&15)
// Each lane's 16 bf16 B operands are 32 contiguous bytes; each block's two
// N-tiles are one contiguous 128 KB region (LDS-stageable with 16B copies).
// ---------------------------------------------------------------------------
__global__ __launch_bounds__(256) void k_prep_w(
    const float* __restrict__ Wn, const float* __restrict__ Wd,
    const float* __restrict__ sdale, const float* __restrict__ alpha,
    __bf16* __restrict__ Bsw)
{
    size_t idx = (size_t)blockIdx.x * 256 + threadIdx.x;   // over NU*NU
    int k = (int)(idx & (NU - 1));   // column j of Wsum (reduction dim)
    int n = (int)(idx >> 11);        // row    i of Wsum (output unit)
    float a = fminf(fmaxf(alpha[0], 0.0f), 1.0f);
    float v = a * fabsf(Wd[idx]) * sdale[k] + (1.0f - a) * Wn[idx];

    int nt = n >> 4, ln = n & 15;
    int kc = k >> 5, k32 = k & 31;
    int lh = k32 >> 4, e = k32 & 15;
    size_t pos = (((size_t)(nt * 64 + kc) * 32) + (lh * 16 + ln)) * 16 + e;
    Bsw[pos] = (__bf16)v;
}

// ---------------------------------------------------------------------------
// Prep 2: zero activity[0], A-fragment buffer[0], barrier; inp_abs = |inp|
// ---------------------------------------------------------------------------
__global__ __launch_bounds__(256) void k_prep_misc(
    float* __restrict__ act0, __bf16* __restrict__ Asw0,
    const float* __restrict__ inp, float* __restrict__ inpAbs,
    unsigned* __restrict__ bar)
{
    int idx = blockIdx.x * 256 + threadIdx.x;   // over NB*NU = 131072
    act0[idx] = 0.0f;
    Asw0[idx] = (__bf16)0.0f;
    if (idx < NI * NU) inpAbs[idx] = fabsf(inp[idx]);
    if (idx == 0) bar[0] = 0u;
}

// ---------------------------------------------------------------------------
// Prep 3: transpose inputs [NB,NI,NT] -> xT [NT, NB*NI] so the per-step x_t
// stage in the persistent kernel is fully coalesced.
// ---------------------------------------------------------------------------
__global__ __launch_bounds__(256) void k_prep_x(
    const float* __restrict__ inputs, float* __restrict__ xT)
{
    int idx = blockIdx.x * 256 + threadIdx.x;   // over NT*NB*NI = 524288
    int t = idx >> 10;            // / (NB*NI)
    int c = idx & 1023;           // b*NI + p
    xT[idx] = inputs[(size_t)c * NT + t];
}

// ---------------------------------------------------------------------------
// Persistent RNN scan. 64 blocks x 256 threads = 512 waves; each wave owns one
// 16x16 tile of the 64x2048 activity (mt = wave&3, local nt = wave>>2) for the
// entire T=512 scan. Weights for the block's two N-tiles live in LDS (128 KB).
// Per step/wave: 64 x v_wmma_f32_16x16x32_bf16 over K=2048, A fragments read
// pre-swizzled from global (L2-resident, written by previous step), B from LDS.
// noise is streamed with non-temporal loads so it never evicts weights from L2.
// Steps are separated by a device-scope atomic grid barrier.
// ---------------------------------------------------------------------------
__global__ __launch_bounds__(256) void k_persist(
    float* __restrict__ act0, float* __restrict__ act1,
    const __bf16* __restrict__ Bsw,
    __bf16* __restrict__ Asw0, __bf16* __restrict__ Asw1,
    const float* __restrict__ inpAbs,
    const float* __restrict__ xT,       // [NT, NB*NI]
    const float* __restrict__ noise,    // [NT, NB, NU]
    unsigned* __restrict__ bar)
{
    extern __shared__ __attribute__((aligned(16))) char smem[];
    __bf16* Bs  = (__bf16*)smem;                          // 2*64*32*16 bf16 = 128 KB
    float*  xsh = (float*)(smem + 2 * 64 * 32 * 16 * 2);  // x_t stage, 4 KB

    const int tid  = threadIdx.x;
    const int lane = tid & 31;
    const int w    = tid >> 5;      // wave in block: 0..7
    const int mt   = w & 3;         // M-tile (batch)
    const int ntl  = w >> 2;        // local N-tile in LDS: 0/1
    const int nt   = blockIdx.x * 2 + ntl;
    const int lm   = lane & 15;
    const int lh   = lane >> 4;

    // ---- stage this block's weight slice into LDS (once) ----
    {
        gv4c g = (gv4c)(Bsw + (size_t)blockIdx.x * (2 * 64 * 32 * 16));
        v4i* s = (v4i*)Bs;
        for (int i = tid; i < (2 * 64 * 32 * 16 * 2) / 16; i += 256) s[i] = g[i];
    }
    __syncthreads();

    // ---- per-lane loop-invariants ----
    const int n = nt * 16 + lm;                 // output-unit column
    float wv[NI];
    {
        gfc iA = (gfc)inpAbs;
#pragma unroll
        for (int p = 0; p < NI; ++p) wv[p] = iA[p * NU + n];
    }

    int    actOff[8];
    size_t aswOff[8];
    {
        // A-fragment swizzle for (m, k=n): k32 = grp*16 + e (lane half = grp)
        int c2 = n >> 5, k32 = n & 31;
        int grp = k32 >> 4, khalf = (k32 >> 3) & 1, rem = k32 & 7;
        int e = k32 & 15;
#pragma unroll
        for (int r = 0; r < 8; ++r) {
            int m = mt * 16 + lh * 8 + r;       // batch row (C/D layout)
            actOff[r] = m * NU + n;
            aswOff[r] = (((size_t)((m >> 4) * 64 + c2) * 32)
                         + (grp * 16 + (m & 15))) * 16 + e;
        }
    }

    const v16bf* Bl = (const v16bf*)Bs + ntl * 64 * 32 + lane;
    const float pOld = 1.0f - 1.0f / 5.0f;
    const float pNew = 1.0f / 5.0f;

    float*  actB[2] = {act0, act1};
    __bf16* aswB[2] = {Asw0, Asw1};
    gfc xTG = (gfc)xT;

    for (int t = 0; t < NT; ++t) {
        gfc  actIn  = (gfc)actB[t & 1];
        gf   actOut = (gf)actB[(t + 1) & 1];
        gbf  aswOut = (gbf)aswB[(t + 1) & 1];
        gv16c Ap = (gv16c)((const v16bf*)aswB[t & 1] + (size_t)mt * 64 * 32 + lane);

        // stage x_t into LDS (coalesced from transposed copy)
        for (int i = tid; i < NB * NI; i += 256) xsh[i] = xTG[(size_t)t * (NB * NI) + i];
        __syncthreads();

        v8f acc = {};
#pragma unroll 4
        for (int kc = 0; kc < 64; ++kc) {
            v16bf av = Ap[kc * 32];             // 2x global_load_b128 (L2)
            v16bf bv = Bl[kc * 32];             // 2x ds_load_b128     (LDS)
            acc = __builtin_amdgcn_wmma_f32_16x16x32_bf16(
                false, av, false, bv, (short)0, acc, false, false);
        }

        gfc np = (gfc)(noise + (size_t)t * NB * NU);
#pragma unroll
        for (int r = 0; r < 8; ++r) {
            int m = mt * 16 + lh * 8 + r;
            float xd = 0.0f;
#pragma unroll
            for (int p = 0; p < NI; ++p) xd = fmaf(wv[p], xsh[m * NI + p], xd);
            float nz = __builtin_nontemporal_load(np + actOff[r]);  // streamed once
            float nv = pOld * actIn[actOff[r]] + pNew * (acc[r] + xd) + nz;
            actOut[actOff[r]] = nv;
            aswOut[aswOff[r]] = (__bf16)fmaxf(nv, 0.0f);
        }

        // ---- grid barrier (release writes, wait for all 64 blocks) ----
        __threadfence();
        __syncthreads();
        if (tid == 0) {
            __hip_atomic_fetch_add(bar, 1u, __ATOMIC_RELEASE, __HIP_MEMORY_SCOPE_AGENT);
            unsigned target = (unsigned)(t + 1) * NBLK;
            while (__hip_atomic_load(bar, __ATOMIC_ACQUIRE, __HIP_MEMORY_SCOPE_AGENT) < target)
                __builtin_amdgcn_s_sleep(2);
        }
        __syncthreads();
    }
}

// ---------------------------------------------------------------------------
// Final: out[b] = act[b,:] @ out_w (LDS tree reduction) + copy activity.
// d_out layout: output[64] then activity[64*2048].
// ---------------------------------------------------------------------------
__global__ __launch_bounds__(256) void k_final(
    const float* __restrict__ act, const float* __restrict__ out_w,
    float* __restrict__ out)
{
    __shared__ float red[256];
    int b = blockIdx.x;
    float s = 0.0f;
    for (int i = threadIdx.x; i < NU; i += 256) {
        float v = act[(size_t)b * NU + i];
        s = fmaf(v, out_w[i], s);
        out[NB + (size_t)b * NU + i] = v;
    }
    red[threadIdx.x] = s;
    __syncthreads();
    for (int off = 128; off > 0; off >>= 1) {
        if (threadIdx.x < off) red[threadIdx.x] += red[threadIdx.x + off];
        __syncthreads();
    }
    if (threadIdx.x == 0) out[b] = red[0];
}

// ---------------------------------------------------------------------------
extern "C" void kernel_launch(void* const* d_in, const int* in_sizes, int n_in,
                              void* d_out, int out_size, void* d_ws, size_t ws_size,
                              hipStream_t stream)
{
    const float* inputs   = (const float*)d_in[0];  // [64,16,512]
    const float* noise    = (const float*)d_in[1];  // [512,64,2048]
    const float* inp      = (const float*)d_in[2];  // [16,2048]
    const float* Wrecnon  = (const float*)d_in[3];  // [2048,2048]
    const float* Wrecdale = (const float*)d_in[4];  // [2048,2048]
    const float* out_w    = (const float*)d_in[5];  // [2048,1]
    const float* alpha    = (const float*)d_in[6];  // [1]
    const float* sdale    = (const float*)d_in[7];  // [2048]

    char* ws = (char*)d_ws;
    constexpr size_t BSW_B = (size_t)NU * NU * 2;       // 8 MB  bf16 weights (swizzled)
    constexpr size_t ASW_B = (size_t)NB * NU * 2;       // 256 KB bf16 A-fragments
    constexpr size_t ACT_B = (size_t)NB * NU * 4;       // 512 KB f32 activity
    constexpr size_t INP_B = (size_t)NI * NU * 4;       // 128 KB |inp|
    constexpr size_t XT_B  = (size_t)NT * NB * NI * 4;  // 2 MB transposed inputs

    __bf16*   Bsw    = (__bf16*)(ws);
    __bf16*   Asw0   = (__bf16*)(ws + BSW_B);
    __bf16*   Asw1   = (__bf16*)(ws + BSW_B + ASW_B);
    float*    act0   = (float*)(ws + BSW_B + 2 * ASW_B);
    float*    act1   = (float*)(ws + BSW_B + 2 * ASW_B + ACT_B);
    float*    inpAbs = (float*)(ws + BSW_B + 2 * ASW_B + 2 * ACT_B);
    float*    xT     = (float*)(ws + BSW_B + 2 * ASW_B + 2 * ACT_B + INP_B);
    unsigned* bar    = (unsigned*)(ws + BSW_B + 2 * ASW_B + 2 * ACT_B + INP_B + XT_B);

    k_prep_w<<<(NU * NU) / 256, 256, 0, stream>>>(Wrecnon, Wrecdale, sdale, alpha, Bsw);
    k_prep_misc<<<(NB * NU) / 256, 256, 0, stream>>>(act0, Asw0, inp, inpAbs, bar);
    k_prep_x<<<(NT * NB * NI) / 256, 256, 0, stream>>>(inputs, xT);

    // 128 KB weight slice + 4 KB x_t stage = 132 KB dynamic LDS (<= 320 KB/WGP)
    size_t smemBytes = (size_t)(2 * 64 * 32 * 16 * 2) + (size_t)(NB * NI * 4);
    k_persist<<<NBLK, 256, smemBytes, stream>>>(act0, act1, Bsw, Asw0, Asw1,
                                                inpAbs, xT, noise, bar);

    // after t = 511, newest activity lives in act0 (512 is even)
    k_final<<<NB, 256, 0, stream>>>(act0, out_w, (float*)d_out);
}
